// ParallaxICBlurModel_87136296501461
// MI455X (gfx1250) — matrix operand order repaired and checked by reading, not compiled, not run on hardware
//
#include <hip/hip_runtime.h>

typedef __attribute__((ext_vector_type(16))) _Float16     v16h;
typedef __attribute__((ext_vector_type(8)))  float        v8f;
typedef __attribute__((ext_vector_type(4)))  unsigned int u32x4;
typedef __attribute__((ext_vector_type(8)))  int          i32x8;
typedef __attribute__((ext_vector_type(4)))  int          i32x4;

#define HH 1024
#define WW 1024
#define KS 31
#define NL 16
#define IW 1054                 // padded input height/width
#define BX 128                  // output pixels per block (x)
#define ROWU 79                 // dwords per staged LDS row (158 halves)
#define CPYU (3 * 31 * ROWU)    // dwords per image copy in LDS (= 7347)
#define WSB_BYTES (93 * 32 * 16 * 2)   // 95232 B of pre-swizzled A-matrices

// ---------------------------------------------------------------------------
// Prep 1: swizzle depthwise weights into this code's contiguous-k 16-bit A-matrix (16x32)
// VGPR layout, one 1 KB block per (channel, kernel-row). Tap k=31 is zero pad.
//   lanes 0-15  : M=lane,    K = 0..15
//   lanes 16-31 : M=lane-16, K = 16..31
// element e (0..15) within a lane -> K = khb + e
// ---------------------------------------------------------------------------
__global__ void pib_prep_weights(const float* __restrict__ w,
                                 _Float16* __restrict__ wsB) {
  int t = blockIdx.x * blockDim.x + threadIdx.x;
  if (t >= 93 * 32) return;
  int b = t >> 5, lane = t & 31;
  int c = b / 31, kr = b % 31;
  int layer = lane & 15;
  int khb = (lane >> 4) ? 16 : 0;
  _Float16* o = wsB + (size_t)t * 16;
#pragma unroll
  for (int e = 0; e < 16; ++e) {
    int k = khb + e;
    float v = (k < KS) ? w[(((c * NL + layer) * KS) + kr) * KS + k] : 0.0f;
    o[e] = (_Float16)v;
  }
}

// ---------------------------------------------------------------------------
// Prep 2: convert the padded image to f16 (TDM source buffer in d_ws).
// ---------------------------------------------------------------------------
__global__ void pib_prep_img(const float* __restrict__ img,
                             _Float16* __restrict__ o, int n) {
  int t = blockIdx.x * blockDim.x + threadIdx.x;
  int stride = gridDim.x * blockDim.x;
  for (int i = t; i < n; i += stride) o[i] = (_Float16)img[i];
}

// ---------------------------------------------------------------------------
// Main fused kernel: implicit-GEMM depthwise conv (f16 WMMA, f32 acc) + alpha
// layer-reduction. Block = 256 threads (8 waves); wave w owns output pixels
// x = blockIdx.x*128 + w*16 .. +15 on row y = blockIdx.y, all 3 channels.
// Image tile staged to LDS by the Tensor Data Mover as two copies (shift 0
// and +1 element) so sliding-window B reads are dword-aligned.
// ---------------------------------------------------------------------------
__global__ __launch_bounds__(256) void pib_blur_wmma(
    const _Float16* __restrict__ wsB,    // 93*32*16 swizzled A blocks
    const _Float16* __restrict__ imgh,   // 3*1054*1054 f16 image
    const float*    __restrict__ alpha,  // (4,16,1024,1024)
    const int*      __restrict__ idxp,
    float*          __restrict__ out) {  // (3,1024,1024)
  __shared__ unsigned int smem[2 * CPYU + 32];

  const int y   = blockIdx.y;
  const int xb  = blockIdx.x * BX;
  const int tid = threadIdx.x;
  const int lane = tid & 31;
  const int wv   = tid >> 5;

  // ---- Wave 0 issues 6 TDM tile loads (2 shifted copies x 3 channels) ----
  if (tid < 32) {
    unsigned int lds0 = (unsigned int)(unsigned long long)(&smem[0]);
#pragma unroll
    for (int cp = 0; cp < 2; ++cp) {
#pragma unroll
      for (int c = 0; c < 3; ++c) {
        unsigned long long ga = (unsigned long long)(const void*)(
            imgh + ((size_t)c * IW * IW + (size_t)y * IW + xb + cp));
        unsigned int lds = lds0 + (unsigned int)(((cp * 3 + c) * 31 * ROWU) * 4);
        unsigned int td0 = (unsigned int)(IW - xb - cp);  // elems left in row
        unsigned int td1 = (unsigned int)(IW - y);        // rows left
        // D# group 0: count=1, lds_addr, 57-bit global_addr, type=2
        u32x4 g0 = {1u, lds, (unsigned int)ga,
                    (unsigned int)((ga >> 32) & 0x1FFFFFFull) | 0x80000000u};
        // D# group 1: data_size=2B, dims, tile 158x31, stride 1054
        i32x8 g1;
        g1[0] = 0x00010000;                                   // data_size=1 (2B)
        g1[1] = (int)((td0 & 0xFFFFu) << 16);                 // tensor_dim0 lo
        g1[2] = (int)((td0 >> 16) | ((td1 & 0xFFFFu) << 16)); // td0 hi | td1 lo
        g1[3] = (int)((td1 >> 16) | (158u << 16));            // tile_dim0=158
        g1[4] = 31;                                           // tile_dim1=31
        g1[5] = IW;                                           // dim0 stride
        g1[6] = 0;
        g1[7] = 0;
        i32x4 gz4 = {0, 0, 0, 0};
        i32x8 gz8 = {0, 0, 0, 0, 0, 0, 0, 0};
        __builtin_amdgcn_tensor_load_to_lds(g0, g1, gz4, gz4, gz8, 0);
      }
    }
    __builtin_amdgcn_s_wait_tensorcnt(0);
  }
  __syncthreads();
  asm volatile("" ::: "memory");  // LDS was written behind the compiler's back

  // ---- B-matrix (32x16, 16-bit) lane addressing into the staged image ----
  //   lanes 0-15 : N=lane,    K=0..15  -> halves s..s+15,  s = wv*16+n
  //   lanes16-31 : N=lane-16, K=16..31 -> halves s..s+15,  s = wv*16+n+16
  // parity-select the shifted copy so every b32 load is dword aligned.
  const int n  = lane & 15;
  const int kb = (lane >> 4) << 4;
  const int s  = wv * 16 + n + kb;
  const int p  = s & 1;
  const unsigned int* bbase = smem + p * CPYU + ((s - p) >> 1);

  v8f acc0 = {}, acc1 = {}, acc2 = {};
  const char* wb = (const char*)wsB;

  for (int kr = 0; kr < KS; ++kr) {
#pragma unroll
    for (int c = 0; c < 3; ++c) {
      const int blk = c * KS + kr;
      v16h a = *(const v16h*)(wb + (size_t)blk * 1024 + lane * 32);
      const unsigned int* bp = bbase + blk * ROWU;
      union { v16h h; unsigned int u[8]; } B;
#pragma unroll
      for (int q = 0; q < 8; ++q) B.u[q] = bp[q];
      v8f& acc = (c == 0) ? acc0 : ((c == 1) ? acc1 : acc2);
      acc = __builtin_amdgcn_wmma_f32_16x16x32_f16(
          false, a, false, B.h, (short)0, acc, false, false);
    }
  }

  // ---- Fused alpha reduction over layers ----
  // D layout: VGPR j -> (layer j, pixel n) lanes 0-15; (layer 8+j, pixel n)
  // lanes 16-31. Reduce over j in-lane, then add across lane halves.
  const int id = *idxp;
  const int x  = xb + wv * 16 + n;
  const int lb = (lane >> 4) << 3;  // layer base: 0 or 8
  const float* ap = alpha + ((((size_t)id * NL + lb) * HH + y) * WW + x);
  float av[8];
#pragma unroll
  for (int j = 0; j < 8; ++j) av[j] = ap[(size_t)j * HH * WW];

#pragma unroll
  for (int c = 0; c < 3; ++c) {
    const v8f& acc = (c == 0) ? acc0 : ((c == 1) ? acc1 : acc2);
    float partial = 0.0f;
#pragma unroll
    for (int j = 0; j < 8; ++j) partial += acc[j] * av[j];
    float sum = partial + __shfl_xor(partial, 16, 32);
    if (lane < 16) out[((size_t)c * HH + y) * WW + x] = sum;
  }
}

// ---------------------------------------------------------------------------
extern "C" void kernel_launch(void* const* d_in, const int* in_sizes, int n_in,
                              void* d_out, int out_size, void* d_ws,
                              size_t ws_size, hipStream_t stream) {
  const float* img   = (const float*)d_in[0];  // (1,3,1054,1054)
  const int*   idx   = (const int*)d_in[1];    // scalar
  const float* wgt   = (const float*)d_in[2];  // (48,1,31,31)
  const float* alpha = (const float*)d_in[3];  // (4,16,1024,1024)
  float* out = (float*)d_out;                  // (1,3,1024,1024)

  _Float16* wsB  = (_Float16*)d_ws;
  _Float16* imgh = (_Float16*)((char*)d_ws + WSB_BYTES);

  pib_prep_weights<<<(93 * 32 + 255) / 256, 256, 0, stream>>>(wgt, wsB);
  const int nimg = 3 * IW * IW;
  pib_prep_img<<<2048, 256, 0, stream>>>(img, imgh, nimg);

  dim3 grid(WW / BX, HH);
  pib_blur_wmma<<<grid, 256, 0, stream>>>(wsB, imgh, alpha, idx, out);
}
